// SAGE_67662914781314
// MI455X (gfx1250) — matrix operand order repaired and checked by reading, not compiled, not run on hardware
//
#include <hip/hip_runtime.h>

typedef __attribute__((ext_vector_type(16))) _Float16 v16h;
typedef __attribute__((ext_vector_type(8)))  float    v8f;
typedef _Float16 f16_t;

#define DH   128
#define KTOT 256
#define BSTR 264   // padded LDS column stride (halves): 528B -> 4-bank step/col

// ---------------------------------------------------------------- utilities
__global__ void fill_zero_f32(float* __restrict__ p, long n) {
    long i = (long)blockIdx.x * blockDim.x + threadIdx.x;
    long stride = (long)gridDim.x * blockDim.x;
    for (; i < n; i += stride) p[i] = 0.0f;
}

__global__ void degree_kernel(const int* __restrict__ dst, float* __restrict__ deg, int nE) {
    int e = blockIdx.x * blockDim.x + threadIdx.x;
    if (e < nE) atomicAdd(&deg[dst[e]], 1.0f);
}

__global__ void rdeg_kernel(float* __restrict__ deg, int n) {
    int i = blockIdx.x * blockDim.x + threadIdx.x;
    if (i < n) deg[i] = 1.0f / fmaxf(deg[i], 1.0f);
}

// Pack [W;R] (K=256) into f16, column-major: wcatT[col*256 + k]
__global__ void prep_weights(const float* __restrict__ w, const float* __restrict__ r,
                             f16_t* __restrict__ wcatT, int src_cols, int ncols) {
    int idx = blockIdx.x * blockDim.x + threadIdx.x;
    if (idx >= ncols * KTOT) return;
    int col = idx >> 8;
    int k   = idx & 255;
    float v = 0.0f;
    if (col < src_cols)
        v = (k < DH) ? w[k * src_cols + col] : r[(k - DH) * src_cols + col];
    wcatT[(size_t)col * KTOT + k] = (f16_t)v;
}

// ---------------------------------------------------------------- scatter (edge gather + atomic add)
__global__ __launch_bounds__(256)
void scatter_kernel(const float* __restrict__ h, const int* __restrict__ srcp,
                    const int* __restrict__ dstp, float* __restrict__ agg, int nE) {
    const int lane = threadIdx.x & 31;
    const int e = blockIdx.x * 8 + (threadIdx.x >> 5);   // one wave per edge
    if (e >= nE) return;
    const int s = srcp[e];
    const int d = dstp[e];
    const float4 v = *(const float4*)(h + (size_t)s * DH + lane * 4);
    float* ap = agg + (size_t)d * DH + lane * 4;
    atomicAdd(ap + 0, v.x);
    atomicAdd(ap + 1, v.y);
    atomicAdd(ap + 2, v.z);
    atomicAdd(ap + 3, v.w);
}

// ------------------------------------------------ async global->LDS helpers
// GVS-mode async copy: LDS[vdst] = MEM[saddr + vaddr]; tracked on ASYNCcnt.
__device__ __forceinline__ void async_copy_b128(unsigned lds_byte_addr,
                                                unsigned glob_byte_off,
                                                const void* sbase) {
    asm volatile("global_load_async_to_lds_b128 %0, %1, %2"
                 :: "v"(lds_byte_addr), "v"(glob_byte_off), "s"(sbase)
                 : "memory");
}

__device__ __forceinline__ void wait_asynccnt0() {
#if __has_builtin(__builtin_amdgcn_s_wait_asynccnt)
    __builtin_amdgcn_s_wait_asynccnt(0);
#else
    asm volatile("s_wait_asynccnt 0x0" ::: "memory");
#endif
}

// ---------------------------------------------------------------- WMMA GEMM
// out[m, n] = sum_k A[m,k] * WcatT[n,k] + bias[n]
// A[m, k<128]  = agg[m,k] * rdeg[m]   (mean aggregation term)
// A[m, k>=128] = xin[m,k-128]         (self term)
// Weight pack is staged once per block into LDS via async LDS-copy
// (ASYNCcnt path), B fragments then come from ds_load_b128.
template <int NT>
__global__ __launch_bounds__(256)
void gemm_kernel(const float* __restrict__ agg, const float* __restrict__ rdeg,
                 const float* __restrict__ xin, const f16_t* __restrict__ wcatT,
                 const float* __restrict__ bias, float* __restrict__ out,
                 int nrows, int out_ld, int ncols_valid) {
    __shared__ f16_t sB[NT * 16 * BSTR];

    // ---- stage weights: NT*16 cols x 256 K halves, 16B chunks, padded stride
    {
        const unsigned ldsBase = (unsigned)(unsigned long long)(const void*)&sB[0];
        const int nchunks = NT * 16 * 32;            // 32 x b128 per column
        for (int c = threadIdx.x; c < nchunks; c += 256) {
            const int col = c >> 5, j = c & 31;
            const unsigned goff = (unsigned)((col * KTOT + j * 8) * 2);
            const unsigned loff = ldsBase + (unsigned)((col * BSTR + j * 8) * 2);
            async_copy_b128(loff, goff, (const void*)wcatT);
        }
        wait_asynccnt0();
    }
    __syncthreads();

    const int lane  = threadIdx.x & 31;
    const int strip = blockIdx.x * 8 + (threadIdx.x >> 5);
    const int m0 = strip * 16;
    if (m0 >= nrows) return;                 // wave-uniform

    int row = m0 + (lane & 15);
    if (row > nrows - 1) row = nrows - 1;    // clamp (reads only)
    const int base8 = (lane >> 4) << 3;      // A: K base 0 / 8 per half-wave
    const int bklo  = (lane >> 4) << 3;      // B: K base 0 / 8 per half-wave
    const float rd = rdeg[row];

    v8f c[NT];
    const v8f vzero = {0, 0, 0, 0, 0, 0, 0, 0};
#pragma unroll
    for (int t = 0; t < NT; ++t) c[t] = vzero;

#pragma unroll
    for (int kb = 0; kb < 8; ++kb) {         // K = 256 in steps of 32
        v16h a;
#pragma unroll
        for (int q = 0; q < 2; ++q) {        // two contiguous 8-wide K runs
            const int kk = kb * 32 + base8 + q * 16;
            const float* ap;
            float sc;
            if (kk < DH) { ap = agg + (size_t)row * DH + kk;        sc = rd;   }
            else         { ap = xin + (size_t)row * DH + (kk - DH); sc = 1.0f; }
            const float4 f0 = *(const float4*)(ap);
            const float4 f1 = *(const float4*)(ap + 4);
            a[q * 8 + 0] = (_Float16)(f0.x * sc);
            a[q * 8 + 1] = (_Float16)(f0.y * sc);
            a[q * 8 + 2] = (_Float16)(f0.z * sc);
            a[q * 8 + 3] = (_Float16)(f0.w * sc);
            a[q * 8 + 4] = (_Float16)(f1.x * sc);
            a[q * 8 + 5] = (_Float16)(f1.y * sc);
            a[q * 8 + 6] = (_Float16)(f1.z * sc);
            a[q * 8 + 7] = (_Float16)(f1.w * sc);
        }
        const int klo = kb * 32 + bklo;
#pragma unroll
        for (int nt = 0; nt < NT; ++nt) {
            const int col = nt * 16 + (lane & 15);
            const f16_t* bs = &sB[col * BSTR + klo];
            union { v16h h; float4 f[2]; } bb;
            bb.f[0] = *(const float4*)(bs);
            bb.f[1] = *(const float4*)(bs + 16);
            c[nt] = __builtin_amdgcn_wmma_f32_16x16x32_f16(
                        false, a, false, bb.h, (short)0, c[nt], false, false);
        }
    }

    const int rbump = (lane >> 4) << 3;      // upper half-wave holds rows +8
#pragma unroll
    for (int nt = 0; nt < NT; ++nt) {
        const int col = nt * 16 + (lane & 15);
#pragma unroll
        for (int i = 0; i < 8; ++i) {
            const int rr = m0 + i + rbump;
            if (col < ncols_valid && rr < nrows)
                out[(size_t)rr * out_ld + col] = c[nt][i] + bias[col];
        }
    }
}

// ---------------------------------------------------------------- batch norm
__global__ __launch_bounds__(128)
void colstats_kernel(const float* __restrict__ h, float* __restrict__ sum,
                     float* __restrict__ sumsq, int nrows) {
    const int col = threadIdx.x;             // 128 threads = 128 columns
    float s = 0.0f, s2 = 0.0f;
    for (int rr = blockIdx.x; rr < nrows; rr += gridDim.x) {
        const float v = h[(size_t)rr * DH + col];
        s += v;
        s2 += v * v;
    }
    atomicAdd(&sum[col], s);
    atomicAdd(&sumsq[col], s2);
}

__global__ __launch_bounds__(128)
void bn_finalize(const float* __restrict__ sum, const float* __restrict__ sumsq,
                 const float* __restrict__ g, const float* __restrict__ be,
                 float* __restrict__ scale, float* __restrict__ shift, float inv_n) {
    const int c = threadIdx.x;
    const float mu  = sum[c] * inv_n;
    const float var = fmaxf(sumsq[c] * inv_n - mu * mu, 0.0f);
    const float sc  = g[c] * rsqrtf(var + 1e-5f);
    scale[c] = sc;
    shift[c] = be[c] - mu * sc;
}

__global__ void bn_apply_relu(float* __restrict__ h, const float* __restrict__ scale,
                              const float* __restrict__ shift, long n) {
    long i = (long)blockIdx.x * blockDim.x + threadIdx.x;
    long stride = (long)gridDim.x * blockDim.x;
    for (; i < n; i += stride) {
        const int c = (int)(i & (DH - 1));
        h[i] = fmaxf(h[i] * scale[c] + shift[c], 0.0f);
    }
}

// ---------------------------------------------------------------- log-softmax (in place)
__global__ void logsoftmax_kernel(float* __restrict__ out, int nrows, int ncols) {
    const int rr = blockIdx.x * blockDim.x + threadIdx.x;
    if (rr >= nrows) return;
    float* p = out + (size_t)rr * ncols;
    float m = -3.4e38f;
    for (int i = 0; i < ncols; ++i) m = fmaxf(m, p[i]);
    float s = 0.0f;
    for (int i = 0; i < ncols; ++i) s += __expf(p[i] - m);
    const float ls = __logf(s) + m;
    for (int i = 0; i < ncols; ++i) p[i] = p[i] - ls;
}

// ---------------------------------------------------------------- host orchestration
static inline size_t align256(size_t v) { return (v + 255) & ~(size_t)255; }

extern "C" void kernel_launch(void* const* d_in, const int* in_sizes, int n_in,
                              void* d_out, int out_size, void* d_ws, size_t ws_size,
                              hipStream_t stream) {
    const float* x   = (const float*)d_in[0];
    const int*   ei  = (const int*)d_in[1];
    const float* w1  = (const float*)d_in[2];
    const float* r1  = (const float*)d_in[3];
    const float* b1  = (const float*)d_in[4];
    const float* g1  = (const float*)d_in[5];
    const float* be1 = (const float*)d_in[6];
    const float* w2  = (const float*)d_in[7];
    const float* r2  = (const float*)d_in[8];
    const float* b2  = (const float*)d_in[9];
    const float* g2  = (const float*)d_in[10];
    const float* be2 = (const float*)d_in[11];
    const float* w3  = (const float*)d_in[12];
    const float* r3  = (const float*)d_in[13];
    const float* b3  = (const float*)d_in[14];

    const int N = in_sizes[0] / DH;          // 100000
    const int E = in_sizes[1] / 2;           // 1600000
    const int DOUT = (int)(in_sizes[12] / DH); // w3 = 128*40 -> 40
    const int* src = ei;
    const int* dst = ei + E;

    char* ws = (char*)d_ws;
    size_t o = 0;
    float* rdeg = (float*)(ws + o); o = align256(o + (size_t)N * 4);
    float* agg  = (float*)(ws + o); o = align256(o + (size_t)N * DH * 4);
    float* h1   = (float*)(ws + o); o = align256(o + (size_t)N * DH * 4);
    float* h2   = (float*)(ws + o); o = align256(o + (size_t)N * DH * 4);
    f16_t* wc1  = (f16_t*)(ws + o); o = align256(o + (size_t)DH * KTOT * 2);
    f16_t* wc2  = (f16_t*)(ws + o); o = align256(o + (size_t)DH * KTOT * 2);
    f16_t* wc3  = (f16_t*)(ws + o); o = align256(o + (size_t)64 * KTOT * 2);
    float* ssum = (float*)(ws + o); o = align256(o + 256 * 4);
    float* scl  = (float*)(ws + o); o = align256(o + 128 * 4);
    float* shf  = (float*)(ws + o); o = align256(o + 128 * 4);
    (void)ws_size; (void)n_in; (void)out_size;

    const long NF = (long)N * DH;
    const int strips  = (N + 15) / 16;
    const int gblocks = (strips + 7) / 8;
    const int eblocks = (E + 7) / 8;

    // degree -> reciprocal
    fill_zero_f32<<<(N + 255) / 256, 256, 0, stream>>>(rdeg, N);
    degree_kernel<<<(E + 255) / 256, 256, 0, stream>>>(dst, rdeg, E);
    rdeg_kernel<<<(N + 255) / 256, 256, 0, stream>>>(rdeg, N);

    // weight packs (f16, col-major K=256)
    prep_weights<<<(DH * KTOT + 255) / 256, 256, 0, stream>>>(w1, r1, wc1, DH, DH);
    prep_weights<<<(DH * KTOT + 255) / 256, 256, 0, stream>>>(w2, r2, wc2, DH, DH);
    prep_weights<<<(64 * KTOT + 255) / 256, 256, 0, stream>>>(w3, r3, wc3, DOUT, 64);

    // ---- layer 1
    fill_zero_f32<<<8192, 256, 0, stream>>>(agg, NF);
    scatter_kernel<<<eblocks, 256, 0, stream>>>(x, src, dst, agg, E);
    gemm_kernel<8><<<gblocks, 256, 0, stream>>>(agg, rdeg, x, wc1, b1, h1, N, DH, DH);
    fill_zero_f32<<<1, 256, 0, stream>>>(ssum, 256);
    colstats_kernel<<<512, 128, 0, stream>>>(h1, ssum, ssum + 128, N);
    bn_finalize<<<1, 128, 0, stream>>>(ssum, ssum + 128, g1, be1, scl, shf, 1.0f / (float)N);
    bn_apply_relu<<<8192, 256, 0, stream>>>(h1, scl, shf, NF);

    // ---- layer 2
    fill_zero_f32<<<8192, 256, 0, stream>>>(agg, NF);
    scatter_kernel<<<eblocks, 256, 0, stream>>>(h1, src, dst, agg, E);
    gemm_kernel<8><<<gblocks, 256, 0, stream>>>(agg, rdeg, h1, wc2, b2, h2, N, DH, DH);
    fill_zero_f32<<<1, 256, 0, stream>>>(ssum, 256);
    colstats_kernel<<<512, 128, 0, stream>>>(h2, ssum, ssum + 128, N);
    bn_finalize<<<1, 128, 0, stream>>>(ssum, ssum + 128, g2, be2, scl, shf, 1.0f / (float)N);
    bn_apply_relu<<<8192, 256, 0, stream>>>(h2, scl, shf, NF);

    // ---- layer 3 (writes straight into d_out, 40 valid columns) + log-softmax
    fill_zero_f32<<<8192, 256, 0, stream>>>(agg, NF);
    scatter_kernel<<<eblocks, 256, 0, stream>>>(h2, src, dst, agg, E);
    gemm_kernel<4><<<gblocks, 256, 0, stream>>>(agg, rdeg, h2, wc3, b3, (float*)d_out, N, DOUT, DOUT);
    logsoftmax_kernel<<<(N + 255) / 256, 256, 0, stream>>>((float*)d_out, N, DOUT);
}
